// Attention_68856915689601
// MI455X (gfx1250) — matrix operand order repaired and checked by reading, not compiled, not run on hardware
//
#include <hip/hip_runtime.h>

#define BB 8
#define NQb 1024
#define SSb 2048
#define DDb 512
#define VROW 516   // padded row stride in dwords for the s-paired V tile (conflict-free)

typedef __attribute__((ext_vector_type(16))) __bf16 v16bf;
typedef __attribute__((ext_vector_type(8)))  float  v8f;
typedef __attribute__((ext_vector_type(4)))  float  v4f;
typedef __attribute__((ext_vector_type(4)))  unsigned int v4u;

union AFrag { v16bf v; v4u q[2]; };
union BFrag { v16bf v; unsigned int w[8]; };

__global__ __launch_bounds__(256)
void Attention_68856915689601_kernel(const float* __restrict__ Q,
                                     const float* __restrict__ K,
                                     const float* __restrict__ V,
                                     const int*   __restrict__ MK,
                                     float* __restrict__ Out)
{
    __shared__ __align__(16) __bf16 Qs[32 * DDb];          // 32 KB  Q tile, bf16
    __shared__ __align__(16) float  Ps[32 * 128];          // 16 KB  raw scores
    __shared__ __align__(16) __bf16 Pb[32 * 128];          //  8 KB  masked probs, bf16
    __shared__ __align__(16) unsigned int Vsp[16 * VROW];  // 33 KB  32s x 512d bf16, s-paired
    __shared__ float redA[32 * 8];
    __shared__ float redB[32 * 8];
    __shared__ float mrow[32];
    __shared__ float lrow[32];
    __shared__ float arow[32];

    const int tid   = threadIdx.x;
    const int wave  = tid >> 5;
    const int lane  = tid & 31;
    const int lhalf = lane >> 4;
    const int lmod  = lane & 15;

    const int b  = blockIdx.x >> 5;          // 32 query-blocks per batch
    const int q0 = (blockIdx.x & 31) << 5;   // 32 queries per WG

    // ---- stage Q (32 x 512 fp32) into LDS as bf16 ----
    {
        const float* qbase = Q + ((size_t)b * NQb + q0) * DDb;
        #pragma unroll
        for (int i = 0; i < 16; ++i) {
            int idx4 = i * 256 + tid;                // 4096 float4's
            v4f t = ((const v4f*)qbase)[idx4];
            __bf16* dst = Qs + idx4 * 4;
            dst[0] = (__bf16)t[0]; dst[1] = (__bf16)t[1];
            dst[2] = (__bf16)t[2]; dst[3] = (__bf16)t[3];
        }
    }
    if (tid < 32) { mrow[tid] = -3.0e38f; lrow[tid] = 0.0f; }

    v8f acc[2][4];                                   // [M-tile][N-tile], 64 VGPRs
    #pragma unroll
    for (int mt = 0; mt < 2; ++mt)
        #pragma unroll
        for (int nt = 0; nt < 4; ++nt)
            #pragma unroll
            for (int v = 0; v < 8; ++v) acc[mt][nt][v] = 0.0f;

    __syncthreads();

    const int r   = tid >> 3;    // phase-2 row (0..31)
    const int sub = tid & 7;     // 8 threads/row, 16 cols each

    #pragma unroll 1
    for (int st = 0; st < SSb / 128; ++st) {
        const int s_base = st * 128;

        // ============ phase 1: S = Q @ K^T  (wave owns 16 key cols, 2 M-tiles) =====
        v8f c0, c1;
        #pragma unroll
        for (int v = 0; v < 8; ++v) { c0[v] = 0.0f; c1[v] = 0.0f; }
        {
            const int skey = s_base + wave * 16 + lmod;       // B: lane -> key (N)
            const float* krow = K + ((size_t)b * SSb + skey) * DDb;
            if (st + 1 < SSb / 128)
                __builtin_prefetch(krow + 128 * DDb, 0, 1);   // next tile's K row
            #pragma unroll
            for (int kk = 0; kk < 16; ++kk) {                 // D = 16 k-steps of 32
                AFrag a0, a1;
                const int ab0 = lmod * DDb + kk * 32 + lhalf * 8;
                a0.q[0] = *(const v4u*)(Qs + ab0);
                a0.q[1] = *(const v4u*)(Qs + ab0 + 16);
                const int ab1 = ab0 + 16 * DDb;
                a1.q[0] = *(const v4u*)(Qs + ab1);
                a1.q[1] = *(const v4u*)(Qs + ab1 + 16);
                v16bf bf;                                     // B: K^T 32x16 cvt->bf16
                const v4f* kp = (const v4f*)(krow + kk * 32 + lhalf * 8);
                #pragma unroll
                for (int q4 = 0; q4 < 4; ++q4) {
                    v4f t = kp[q4 + (q4 & 2)];
                    bf[q4 * 4 + 0] = (__bf16)t[0]; bf[q4 * 4 + 1] = (__bf16)t[1];
                    bf[q4 * 4 + 2] = (__bf16)t[2]; bf[q4 * 4 + 3] = (__bf16)t[3];
                }
                c0 = __builtin_amdgcn_wmma_f32_16x16x32_bf16(false, a0.v, false, bf,
                                                             (short)0, c0, false, false);
                c1 = __builtin_amdgcn_wmma_f32_16x16x32_bf16(false, a1.v, false, bf,
                                                             (short)0, c1, false, false);
            }
        }
        #pragma unroll
        for (int v = 0; v < 8; ++v) {
            Ps[(v + 8 * lhalf) * 128 + wave * 16 + lmod]        = c0[v];
            Ps[(16 + v + 8 * lhalf) * 128 + wave * 16 + lmod]   = c1[v];
        }

        __syncthreads();

        // ============ phase 2: online masked softmax (32 rows x 128 cols) ==========
        {
            const float* prow = Ps + r * 128 + sub * 16;
            float sv[16];
            float mloc = -3.0e38f;
            #pragma unroll
            for (int j4 = 0; j4 < 4; ++j4) {
                v4f t = ((const v4f*)prow)[j4];
                #pragma unroll
                for (int j = 0; j < 4; ++j) {
                    sv[j4 * 4 + j] = t[j];
                    mloc = fmaxf(mloc, t[j]);
                }
            }
            redA[r * 8 + sub] = mloc;
            __syncthreads();
            float tmax = -3.0e38f;
            #pragma unroll
            for (int k = 0; k < 8; ++k) tmax = fmaxf(tmax, redA[r * 8 + k]);
            const float mold = mrow[r];
            const float mnew = fmaxf(mold, tmax);
            const float al   = __expf(mold - mnew);
            const v4u* mp4 = (const v4u*)(MK + (((size_t)b * NQb + q0 + r) * SSb
                                                + s_base + sub * 16));
            float psum = 0.0f;
            __bf16* pbrow = Pb + r * 128 + sub * 16;
            #pragma unroll
            for (int j4 = 0; j4 < 4; ++j4) {
                v4u m = mp4[j4];
                #pragma unroll
                for (int j = 0; j < 4; ++j) {
                    float p = __expf(sv[j4 * 4 + j] - mnew) * (float)(int)m[j];
                    psum += p;
                    pbrow[j4 * 4 + j] = (__bf16)p;
                }
            }
            redB[r * 8 + sub] = psum;
            __syncthreads();
            if (sub == 0) {
                float ts = 0.0f;
                #pragma unroll
                for (int k = 0; k < 8; ++k) ts += redB[r * 8 + k];
                lrow[r] = lrow[r] * al + ts;
                mrow[r] = mnew;
                arow[r] = al;
            }
        }
        __syncthreads();

        // ============ phase 3: O = O*alpha + P @ V  (wave owns 64 D cols) ==========
        {
            float as0[8], as1[8];
            #pragma unroll
            for (int v = 0; v < 8; ++v) {
                as0[v] = arow[v + 8 * lhalf];
                as1[v] = arow[16 + v + 8 * lhalf];
            }
            #pragma unroll
            for (int nt = 0; nt < 4; ++nt)
                #pragma unroll
                for (int v = 0; v < 8; ++v) {
                    acc[0][nt][v] *= as0[v];
                    acc[1][nt][v] *= as1[v];
                }

            __bf16* Vh = (__bf16*)Vsp;
            #pragma unroll 1
            for (int kk = 0; kk < 4; ++kk) {
                // --- cooperatively stage V slab (32 s x 512 d) as bf16, s-paired ---
                {
                    const float* vbase = V + ((size_t)b * SSb + s_base + kk * 32) * DDb;
                    #pragma unroll
                    for (int i = 0; i < 16; ++i) {
                        int idx4 = i * 256 + tid;            // 4096 float4's
                        v4f t = ((const v4f*)vbase)[idx4];
                        int flat = idx4 * 4;
                        int sl = flat >> 9;                  // s row in slab
                        int dc = flat & 511;                 // d col
                        // element (s,d) -> Vsp dword (s/2)*VROW + d, half (s&1)
                        __bf16* dst = Vh + ((size_t)((sl >> 1) * VROW + dc) * 2 + (sl & 1));
                        dst[0] = (__bf16)t[0]; dst[2] = (__bf16)t[1];
                        dst[4] = (__bf16)t[2]; dst[6] = (__bf16)t[3];
                    }
                }
                __syncthreads();

                AFrag a0, a1;                                // A: P 16x32 bf16 from LDS
                const int pb0 = lmod * 128 + kk * 32 + lhalf * 8;
                a0.q[0] = *(const v4u*)(Pb + pb0);
                a0.q[1] = *(const v4u*)(Pb + pb0 + 16);
                const int pb1 = pb0 + 16 * 128;
                a1.q[0] = *(const v4u*)(Pb + pb1);
                a1.q[1] = *(const v4u*)(Pb + pb1 + 16);

                #pragma unroll
                for (int nt = 0; nt < 4; ++nt) {
                    const int dcol = wave * 64 + nt * 16 + lmod;   // B: V 32x16
                    const unsigned int* vrow = Vsp + lhalf * 4 * VROW + dcol;
                    BFrag bf;                                 // 8 x ds_load_b32
                    #pragma unroll
                    for (int j = 0; j < 8; ++j) bf.w[j] = vrow[(size_t)(((j & 4) << 1) + (j & 3)) * VROW];
                    acc[0][nt] = __builtin_amdgcn_wmma_f32_16x16x32_bf16(
                        false, a0.v, false, bf.v, (short)0, acc[0][nt], false, false);
                    acc[1][nt] = __builtin_amdgcn_wmma_f32_16x16x32_bf16(
                        false, a1.v, false, bf.v, (short)0, acc[1][nt], false, false);
                }
                __syncthreads();   // protect Vsp reuse by next kk / next S-tile
            }
        }
    }

    // ---- normalize by running sum and store fp32 ----
    float li0[8], li1[8];
    #pragma unroll
    for (int v = 0; v < 8; ++v) {
        li0[v] = 1.0f / lrow[v + 8 * lhalf];
        li1[v] = 1.0f / lrow[16 + v + 8 * lhalf];
    }
    #pragma unroll
    for (int nt = 0; nt < 4; ++nt) {
        const int dcol = wave * 64 + nt * 16 + lmod;
        float* op0 = Out + ((size_t)b * NQb + q0 + 8 * lhalf) * DDb + dcol;
        float* op1 = Out + ((size_t)b * NQb + q0 + 16 + 8 * lhalf) * DDb + dcol;
        #pragma unroll
        for (int v = 0; v < 8; ++v) {
            op0[(size_t)v * DDb] = acc[0][nt][v] * li0[v];
            op1[(size_t)v * DDb] = acc[1][nt][v] * li1[v];
        }
    }
}

extern "C" void kernel_launch(void* const* d_in, const int* in_sizes, int n_in,
                              void* d_out, int out_size, void* d_ws, size_t ws_size,
                              hipStream_t stream) {
    (void)in_sizes; (void)n_in; (void)out_size; (void)d_ws; (void)ws_size;
    const float* Q  = (const float*)d_in[0];
    const float* K  = (const float*)d_in[1];
    const float* V  = (const float*)d_in[2];
    const int*   MKp = (const int*)d_in[3];
    float* O = (float*)d_out;
    dim3 grid(BB * (NQb / 32));   // 256 workgroups
    dim3 block(256);              // 8 wave32 waves
    Attention_68856915689601_kernel<<<grid, block, 0, stream>>>(Q, K, V, MKp, O);
}
